// PointNet2ClassificationSSG_39015482917217
// MI455X (gfx1250) — matrix operand order repaired and checked by reading, not compiled, not run on hardware
//
#include <hip/hip_runtime.h>

// ---------------------------------------------------------------------------
// Types for CDNA5 WMMA (gfx1250, wave32)
// ---------------------------------------------------------------------------
typedef __attribute__((ext_vector_type(16))) __bf16 v16bf;
typedef __attribute__((ext_vector_type(8)))  float  v8f;
typedef __attribute__((ext_vector_type(4)))  int    v4i;

// Address-space-qualified pointers for the async global->LDS copy builtin:
//   param0: v4i in addrspace(1) (global), param1: v4i in addrspace(3) (LDS)
typedef __attribute__((address_space(1))) v4i* g4p;
typedef __attribute__((address_space(3))) v4i* l4p;

#define INV_BN 0.99999500003749977f   // 1/sqrt(1+1e-5)

#if defined(__has_builtin)
#  if __has_builtin(__builtin_amdgcn_global_load_async_to_lds_b128)
#    define HAVE_ASYNC_LDS 1
#  else
#    define HAVE_ASYNC_LDS 0
#  endif
#  if __has_builtin(__builtin_amdgcn_s_wait_asynccnt)
#    define HAVE_WAIT_ASYNC 1
#  else
#    define HAVE_WAIT_ASYNC 0
#  endif
#else
#  define HAVE_ASYNC_LDS 0
#  define HAVE_WAIT_ASYNC 0
#endif

static __device__ __forceinline__ __bf16 f2bf(float f) { return (__bf16)f; }

// ---------------------------------------------------------------------------
// Farthest point sampling: one block per batch. dists lives in LDS.
// ---------------------------------------------------------------------------
__global__ __launch_bounds__(256)
void fps_kernel(const float* __restrict__ pts, int stride, int N, int npoint,
                int* __restrict__ out_idx, float* __restrict__ new_xyz)
{
    __shared__ float dists[4096];
    __shared__ float rv[256];
    __shared__ int   ri[256];
    __shared__ int   s_far;

    const int b   = blockIdx.x;
    const int tid = threadIdx.x;
    const float* base = pts + (size_t)b * N * stride;

    for (int i = tid; i < N; i += 256) dists[i] = 1e10f;
    __syncthreads();

    int far = 0;
    for (int it = 0; it < npoint; ++it) {
        if (tid == 0) out_idx[(size_t)b * npoint + it] = far;
        const float cx = base[(size_t)far * stride + 0];
        const float cy = base[(size_t)far * stride + 1];
        const float cz = base[(size_t)far * stride + 2];

        float bestv = -1.0f; int besti = 0;
        for (int i = tid; i < N; i += 256) {
            const float dx = base[(size_t)i * stride + 0] - cx;
            const float dy = base[(size_t)i * stride + 1] - cy;
            const float dz = base[(size_t)i * stride + 2] - cz;
            const float d  = dx * dx + dy * dy + dz * dz;
            const float m  = fminf(dists[i], d);
            dists[i] = m;
            if (m > bestv) { bestv = m; besti = i; }
        }
        rv[tid] = bestv; ri[tid] = besti;
        __syncthreads();
        for (int off = 128; off > 0; off >>= 1) {
            if (tid < off) {
                if (rv[tid + off] > rv[tid] ||
                    (rv[tid + off] == rv[tid] && ri[tid + off] < ri[tid])) {
                    rv[tid] = rv[tid + off]; ri[tid] = ri[tid + off];
                }
            }
            __syncthreads();
        }
        if (tid == 0) s_far = ri[0];
        __syncthreads();
        far = s_far;
        __syncthreads();
    }

    for (int i = tid; i < npoint; i += 256) {
        const int id = out_idx[(size_t)b * npoint + i];
        new_xyz[((size_t)b * npoint + i) * 3 + 0] = base[(size_t)id * stride + 0];
        new_xyz[((size_t)b * npoint + i) * 3 + 1] = base[(size_t)id * stride + 1];
        new_xyz[((size_t)b * npoint + i) * 3 + 2] = base[(size_t)id * stride + 2];
    }
}

// ---------------------------------------------------------------------------
// Ball query + grouping: one wave32 per query point.  Ordered selection of the
// first `nsample` in-radius indices via __ballot (matches reference top-k of
// smallest indices).  Writes K-padded bf16 rows [rel_xyz(3) | feats(fc) | 0...].
// ---------------------------------------------------------------------------
__global__ __launch_bounds__(256)
void group_kernel(const float* __restrict__ xyz, int xs,
                  const float* __restrict__ feat, int fs, int fc,
                  const float* __restrict__ new_xyz,
                  int N, int S, int nsample, float r2,
                  __bf16* __restrict__ A, int Kpad)
{
    __shared__ int sel[8][128];
    const int b    = blockIdx.y;
    const int wave = threadIdx.x >> 5;
    const int lane = threadIdx.x & 31;
    const int s    = blockIdx.x * 8 + wave;
    if (s >= S) return;

    const float* xb = xyz  + (size_t)b * N * xs;
    const float* fb = feat + (size_t)b * N * fs;
    const float qx = new_xyz[((size_t)b * S + s) * 3 + 0];
    const float qy = new_xyz[((size_t)b * S + s) * 3 + 1];
    const float qz = new_xyz[((size_t)b * S + s) * 3 + 2];

    int cnt = 0;
    for (int basei = 0; basei < N && cnt < nsample; basei += 32) {
        const int i = basei + lane;
        bool ok = false;
        if (i < N) {
            const float dx = xb[(size_t)i * xs + 0] - qx;
            const float dy = xb[(size_t)i * xs + 1] - qy;
            const float dz = xb[(size_t)i * xs + 2] - qz;
            ok = (dx * dx + dy * dy + dz * dz) <= r2;
        }
        const unsigned long long mask = __ballot(ok);
        const int pos = cnt + __popcll(mask & ((1ull << lane) - 1ull));
        if (ok && pos < nsample) sel[wave][pos] = i;
        cnt += __popcll(mask);
    }
    if (cnt > nsample) cnt = nsample;
    asm volatile("s_wait_dscnt 0" ::: "memory");   // cross-lane LDS visibility
    const int first = (cnt > 0) ? sel[wave][0] : 0;

    for (int j = lane; j < nsample; j += 32) {
        const int id = (j < cnt) ? sel[wave][j] : first;
        __bf16* row = A + ((size_t)((size_t)b * S + s) * nsample + j) * Kpad;
        row[0] = f2bf(xb[(size_t)id * xs + 0] - qx);
        row[1] = f2bf(xb[(size_t)id * xs + 1] - qy);
        row[2] = f2bf(xb[(size_t)id * xs + 2] - qz);
        for (int c = 0; c < fc; ++c)
            row[3 + c] = f2bf(fb[(size_t)id * fs + c]);
        for (int c = 3 + fc; c < Kpad; ++c)
            row[c] = f2bf(0.0f);
    }
}

// ---------------------------------------------------------------------------
// "Group-all" for SA3: concat xyz(3) + feats(C) into K-padded bf16 rows.
// ---------------------------------------------------------------------------
__global__ __launch_bounds__(256)
void group_all_kernel(const float* __restrict__ xyz, const float* __restrict__ feats,
                      int C, __bf16* __restrict__ A, int Kpad)
{
    const int row = blockIdx.x;
    __bf16* dst = A + (size_t)row * Kpad;
    for (int c = threadIdx.x; c < Kpad; c += blockDim.x) {
        float v = 0.0f;
        if (c < 3)            v = xyz[(size_t)row * 3 + c];
        else if (c < 3 + C)   v = feats[(size_t)row * C + (c - 3)];
        dst[c] = f2bf(v);
    }
}

// ---------------------------------------------------------------------------
// Weight prep: f32 W[O][C]  ->  bf16 Wp[Kpad][Opad]  (Wp[k][o] = W[o][k])
// ---------------------------------------------------------------------------
__global__ __launch_bounds__(256)
void prep_w(const float* __restrict__ w, __bf16* __restrict__ out,
            int O, int C, int Kpad, int Opad)
{
    const int i = blockIdx.x * 256 + threadIdx.x;
    if (i >= Kpad * Opad) return;
    const int k = i / Opad, o = i % Opad;
    const float v = (k < C && o < O) ? w[(size_t)o * C + k] : 0.0f;
    out[i] = f2bf(v);
}

// ---------------------------------------------------------------------------
// Fused GEMM + BN-scale + bias + ReLU epilogue using v_wmma_f32_16x16x32_bf16.
// Block = 256 threads = 8 waves; each wave owns one 16(M)x32(O) tile (two
// accumulators sharing one A fragment -> half the A traffic per output).
// The 32-wide B panel for the whole K range is staged once into LDS, using
// GLOBAL_LOAD_ASYNC_TO_LDS_B128 when the toolchain exposes it.
// ---------------------------------------------------------------------------
__global__ __launch_bounds__(256)
void gemm_kernel(const __bf16* __restrict__ A, const __bf16* __restrict__ Bw,
                 const float* __restrict__ g, const float* __restrict__ bias,
                 __bf16* __restrict__ outb, float* __restrict__ outf,
                 int M, int Kpad, int Opad, int Oreal, int do_relu)
{
    __shared__ __align__(32) __bf16 Bs[1024 * 32];   // up to K=1024, 64 KB

    const int tid   = threadIdx.x;
    const int lane  = tid & 31;
    const int wave  = tid >> 5;
    const int obase = blockIdx.y * 32;

    // Stage B panel (Kpad x 32) into LDS: one 64B row per k.
    for (int k = tid; k < Kpad; k += 256) {
        const __bf16* src = Bw + (size_t)k * Opad + obase;
        __bf16*       dst = &Bs[k * 32];
#if HAVE_ASYNC_LDS
        #pragma unroll
        for (int j = 0; j < 4; ++j)
            __builtin_amdgcn_global_load_async_to_lds_b128(
                (g4p)(const_cast<__bf16*>(src) + j * 8),
                (l4p)(dst + j * 8),
                0, 0);
#else
        const uint4* s4 = reinterpret_cast<const uint4*>(src);
        uint4*       d4 = reinterpret_cast<uint4*>(dst);
        d4[0] = s4[0]; d4[1] = s4[1]; d4[2] = s4[2]; d4[3] = s4[3];
#endif
    }
#if HAVE_ASYNC_LDS
#  if HAVE_WAIT_ASYNC
    __builtin_amdgcn_s_wait_asynccnt(0);
#  else
    asm volatile("s_wait_asynccnt 0" ::: "memory");
#  endif
#endif
    __syncthreads();

    const int tileM = blockIdx.x * 128 + wave * 16;
    if (tileM >= M) return;

    const int rowA = tileM + (lane & 15);
    const int klo  = (lane < 16) ? 0 : 8;          // 16-bit A-matrix lane layout
    const __bf16* arow = A + (size_t)rowA * Kpad + klo;

    union Frag { v16bf v; uint4 u[2]; };
    v8f acc[2] = {};

    for (int k0 = 0; k0 < Kpad; k0 += 32) {
        Frag af, b0, b1;
        af.u[0] = *reinterpret_cast<const uint4*>(arow + k0);
        af.u[1] = *reinterpret_cast<const uint4*>(arow + k0 + 16);
        if (k0 + 64 < Kpad) __builtin_prefetch(arow + k0 + 64, 0, 1);

        #pragma unroll
        for (int s = 0; s < 16; ++s) {
            const int kk = k0 + klo + s + ((s & 8) ? 8 : 0);
            b0.v[s] = Bs[kk * 32 + (lane & 15)];
            b1.v[s] = Bs[kk * 32 + 16 + (lane & 15)];
        }

        acc[0] = __builtin_amdgcn_wmma_f32_16x16x32_bf16(
                     false, af.v, false, b0.v, (short)0, acc[0], false, false);
        acc[1] = __builtin_amdgcn_wmma_f32_16x16x32_bf16(
                     false, af.v, false, b1.v, (short)0, acc[1], false, false);
    }

    // Epilogue: y = relu(acc * (INV*g) + b) for both 16-wide column halves.
    #pragma unroll
    for (int half = 0; half < 2; ++half) {
        const int n = obase + half * 16 + (lane & 15);
        float sc = 1.0f, bi = 0.0f;
        if (n < Oreal) {
            if (g)    sc = INV_BN * g[n];
            if (bias) bi = bias[n];
        }
        #pragma unroll
        for (int r = 0; r < 8; ++r) {
            const int m = tileM + r + ((lane < 16) ? 0 : 8);
            float v = acc[half][r] * sc + bi;
            if (do_relu) v = fmaxf(v, 0.0f);
            if (n < Oreal && m < M) {
                if (outb) outb[(size_t)m * Opad + n] = f2bf(v);
                if (outf) outf[(size_t)m * Opad + n] = v;
            }
        }
    }
}

// ---------------------------------------------------------------------------
// Max-pool over the sample dimension: in bf16 [R, n, C] -> f32 (+opt bf16) [R, C]
// ---------------------------------------------------------------------------
__global__ __launch_bounds__(256)
void maxpool_kernel(const __bf16* __restrict__ in, int n, int C,
                    float* __restrict__ outf, __bf16* __restrict__ outb)
{
    const int r = blockIdx.x;
    for (int c = threadIdx.x; c < C; c += blockDim.x) {
        const __bf16* p = in + (size_t)r * n * C + c;
        float m = (float)p[0];
        for (int j = 1; j < n; ++j) m = fmaxf(m, (float)p[(size_t)j * C]);
        if (outf) outf[(size_t)r * C + c] = m;
        if (outb) outb[(size_t)r * C + c] = f2bf(m);
    }
}

__global__ void copy_out_kernel(const float* __restrict__ src, float* __restrict__ dst,
                                int rows, int cols, int sstride)
{
    const int i = blockIdx.x * blockDim.x + threadIdx.x;
    if (i < rows * cols) {
        const int r = i / cols, c = i % cols;
        dst[i] = src[(size_t)r * sstride + c];
    }
}

// ---------------------------------------------------------------------------
// Host orchestration
// ---------------------------------------------------------------------------
struct LayerSpec { int wi, gi, bi, C, O, Kpad, Opad; };

extern "C" void kernel_launch(void* const* d_in, const int* in_sizes, int n_in,
                              void* d_out, int out_size, void* d_ws, size_t ws_size,
                              hipStream_t stream)
{
    (void)in_sizes; (void)n_in; (void)out_size; (void)ws_size;

    const float* pc = (const float*)d_in[0];   // [16, 4096, 6]
    const int B = 16, N = 4096;

    // ---- workspace bump allocator (256B aligned) ----
    char* ws = (char*)d_ws;
    size_t off = 0;
    auto alloc = [&](size_t bytes) -> void* {
        void* p = ws + off;
        off = (off + bytes + 255) & ~(size_t)255;
        return p;
    };

    int*    idx1 = (int*)   alloc((size_t)B * 512 * 4);
    float*  nx1  = (float*) alloc((size_t)B * 512 * 3 * 4);
    int*    idx2 = (int*)   alloc((size_t)B * 128 * 4);
    float*  nx2  = (float*) alloc((size_t)B * 128 * 3 * 4);
    float*  F1   = (float*) alloc((size_t)B * 512 * 128 * 4);
    float*  F2   = (float*) alloc((size_t)B * 128 * 256 * 4);
    float*  F3   = (float*) alloc((size_t)B * 1024 * 4);
    __bf16* F3b  = (__bf16*)alloc((size_t)B * 1024 * 2);
    __bf16* fc1o = (__bf16*)alloc((size_t)B * 512 * 2);
    __bf16* fc2o = (__bf16*)alloc((size_t)B * 256 * 2);
    float*  fc3o = (float*) alloc((size_t)B * 64 * 4);

    static const LayerSpec LS[12] = {
        { 1,  2,  3,    6,   64,   32,   64},   // sa1.0
        { 4,  5,  6,   64,   64,   64,   64},   // sa1.1
        { 7,  8,  9,   64,  128,   64,  128},   // sa1.2
        {10, 11, 12,  131,  128,  160,  128},   // sa2.0
        {13, 14, 15,  128,  128,  128,  128},   // sa2.1
        {16, 17, 18,  128,  256,  128,  256},   // sa2.2
        {19, 20, 21,  259,  256,  288,  256},   // sa3.0
        {22, 23, 24,  256,  512,  256,  512},   // sa3.1
        {25, 26, 27,  512, 1024,  512, 1024},   // sa3.2
        {28, 29, 30, 1024,  512, 1024,  512},   // fc1
        {31, 32, 33,  512,  256,  512,  256},   // fc2
        {34, -1, 35,  256,   40,  256,   64},   // fc3
    };

    __bf16* Wp[12];
    for (int i = 0; i < 12; ++i) {
        const size_t elems = (size_t)LS[i].Kpad * LS[i].Opad;
        Wp[i] = (__bf16*)alloc(elems * 2);
        const int total = (int)elems;
        prep_w<<<(total + 255) / 256, 256, 0, stream>>>(
            (const float*)d_in[LS[i].wi], Wp[i], LS[i].O, LS[i].C, LS[i].Kpad, LS[i].Opad);
    }

    __bf16* P0 = (__bf16*)alloc((size_t)134217728);   // 128 MB ping
    __bf16* P1 = (__bf16*)alloc((size_t)134217728);   // 128 MB pong

    auto gemm = [&](const __bf16* Amat, int li, __bf16* outb, float* outf,
                    int M, int relu) {
        const LayerSpec& l = LS[li];
        const float* g  = (l.gi >= 0) ? (const float*)d_in[l.gi] : nullptr;
        const float* bb = (l.bi >= 0) ? (const float*)d_in[l.bi] : nullptr;
        dim3 grid((M + 127) / 128, l.Opad / 32);
        gemm_kernel<<<grid, 256, 0, stream>>>(Amat, Wp[li], g, bb, outb, outf,
                                              M, l.Kpad, l.Opad, l.O, relu);
    };

    // =================== SA1 ===================
    fps_kernel<<<B, 256, 0, stream>>>(pc, 6, N, 512, idx1, nx1);
    group_kernel<<<dim3(512 / 8, B), 256, 0, stream>>>(
        pc, 6, pc + 3, 6, 3, nx1, N, 512, 64, 0.2f * 0.2f, P1, 32);
    const int M1 = B * 512 * 64;
    gemm(P1, 0, P0, nullptr, M1, 1);
    gemm(P0, 1, P1, nullptr, M1, 1);
    gemm(P1, 2, P0, nullptr, M1, 1);
    maxpool_kernel<<<B * 512, 256, 0, stream>>>(P0, 64, 128, F1, nullptr);

    // =================== SA2 ===================
    fps_kernel<<<B, 256, 0, stream>>>(nx1, 3, 512, 128, idx2, nx2);
    group_kernel<<<dim3(128 / 8, B), 256, 0, stream>>>(
        nx1, 3, F1, 128, 128, nx2, 512, 128, 128, 0.4f * 0.4f, P1, 160);
    const int M2 = B * 128 * 128;
    gemm(P1, 3, P0, nullptr, M2, 1);
    gemm(P0, 4, P1, nullptr, M2, 1);
    gemm(P1, 5, P0, nullptr, M2, 1);
    maxpool_kernel<<<B * 128, 256, 0, stream>>>(P0, 128, 256, F2, nullptr);

    // =================== SA3 (group all) ===================
    group_all_kernel<<<B * 128, 256, 0, stream>>>(nx2, F2, 256, P1, 288);
    const int M3 = B * 128;
    gemm(P1, 6, P0, nullptr, M3, 1);
    gemm(P0, 7, P1, nullptr, M3, 1);
    gemm(P1, 8, P0, nullptr, M3, 1);
    maxpool_kernel<<<B, 256, 0, stream>>>(P0, 128, 1024, F3, F3b);

    // =================== FC head ===================
    gemm(F3b, 9,  fc1o, nullptr, B, 1);
    gemm(fc1o, 10, fc2o, nullptr, B, 1);
    gemm(fc2o, 11, nullptr, fc3o, B, 0);

    copy_out_kernel<<<(B * 40 + 255) / 256, 256, 0, stream>>>(
        fc3o, (float*)d_out, B, 40, 64);
}
